// GIKT_53137335386866
// MI455X (gfx1250) — compile-verified, matrix operand order checked by reading, not run on hardware
//
#include <hip/hip_runtime.h>

typedef __attribute__((ext_vector_type(16))) _Float16 v16h;
typedef __attribute__((ext_vector_type(8)))  float    v8f;

static const int cN = 42000, cNUME = 42000, cM = 400000, cNNZ = 500000;
static const int cB = 64, cS = 200, cBS = 12800;
#define NEGB (-1.0e9f)
#define EPSB (1e-5f)

// ------------------------------------------------------------------
// wave helpers (wave32)
// ------------------------------------------------------------------
static __device__ __forceinline__ float wred_sum(float v) {
#pragma unroll
  for (int m = 16; m; m >>= 1) v += __shfl_xor(v, m, 32);
  return v;
}
static __device__ __forceinline__ float wred_max(float v) {
#pragma unroll
  for (int m = 16; m; m >>= 1) v = fmaxf(v, __shfl_xor(v, m, 32));
  return v;
}

// ------------------------------------------------------------------
// Async global -> LDS staging (CDNA5, ASYNCcnt tracked).
// Per-lane 16-byte moves; global/LDS 16B-aligned chunks.
// ------------------------------------------------------------------
static __device__ __forceinline__ void async_chunk16(unsigned int lds, const void* gp) {
  asm volatile("global_load_async_to_lds_b128 %0, %1, off"
               :: "v"(lds), "v"(gp)
               : "memory");
}
static __device__ __forceinline__ void async_stage_f32(const float* __restrict__ g,
                                                       void* ldsdst, int nfloats,
                                                       int tid, int nthreads) {
  unsigned int lds = (unsigned int)(size_t)ldsdst + (unsigned int)tid * 16u;
  const char* gp = (const char*)g + (size_t)tid * 16;
  int iters = nfloats / (nthreads * 4);
  for (int i = 0; i < iters; ++i) {
    async_chunk16(lds, gp);
    lds += (unsigned int)nthreads * 16u;
    gp += (size_t)nthreads * 16;
  }
}
static __device__ __forceinline__ void wait_async0() {
  asm volatile("s_wait_asynccnt 0" ::: "memory");
}

// ------------------------------------------------------------------
// WMMA fragment loaders (CDNA5 16x16x32 f16, wave32 layouts, ISA 7.12.2)
// A (16xK f16 tile in LDS, row-major, ld = K): lane&15 = row M,
//   lane>=16 -> khalf=8 ; element j -> k = kbase + khalf + (j>>3)*16 + (j&7)
// B from W stored [N][K] row-major f16 in LDS (B[k][n] = W[n][k]).
// ------------------------------------------------------------------
static __device__ __forceinline__ v16h frag_a(const _Float16* x, int ld, int kbase, int lane) {
  int r = lane & 15;
  int khalf = (lane & 16) >> 1;  // 0 or 8
  const _Float16* p = x + r * ld + kbase + khalf;
  v16h a;
#pragma unroll
  for (int j = 0; j < 16; ++j) a[j] = p[((j >> 3) << 4) + (j & 7)];
  return a;
}
static __device__ __forceinline__ v16h frag_b(const _Float16* Wh, int ldk, int nbase, int kbase, int lane) {
  int n = nbase + (lane & 15);
  int khalf = (lane & 16) >> 1;
  const _Float16* p = Wh + n * ldk + kbase + khalf;
  v16h b;
#pragma unroll
  for (int j = 0; j < 16; ++j) b[j] = p[((j >> 3) << 4) + (j & 7)];
  return b;
}
static __device__ __forceinline__ void zero4(v8f a[4]) {
#pragma unroll
  for (int nt = 0; nt < 4; ++nt)
#pragma unroll
    for (int e = 0; e < 8; ++e) a[nt][e] = 0.f;
}
// acc[nt] covers output cols nbase4+nt*16 .. +15, rows = 16-row tile
static __device__ __forceinline__ void gemm16(const _Float16* xt, const _Float16* Wh, int K,
                                              int nbase4, int lane, v8f acc[4]) {
  for (int kc = 0; kc < K; kc += 32) {
    v16h a = frag_a(xt, K, kc, lane);
#pragma unroll
    for (int nt = 0; nt < 4; ++nt) {
      v16h b = frag_b(Wh, K, nbase4 + nt * 16, kc, lane);
      acc[nt] = __builtin_amdgcn_wmma_f32_16x16x32_f16(false, a, false, b, (short)0, acc[nt], false, false);
    }
  }
}
// LayerNorm of a 16 x din LDS tile -> f16 tile (2 lanes per row: halves)
static __device__ __forceinline__ void ln_tile(const float* raw, _Float16* dst, int din,
                                               const float* g, const float* bb, int lane) {
  int r = lane & 15, half = lane >> 4, hh = din >> 1;
  const float* p = raw + r * din + half * hh;
  float s = 0.f, q = 0.f;
  for (int j = 0; j < hh; ++j) { float v = p[j]; s += v; q += v * v; }
  s += __shfl_xor(s, 16, 32);
  q += __shfl_xor(q, 16, 32);
  float mu = s / (float)din;
  float rs = rsqrtf(q / (float)din - mu * mu + EPSB);
  _Float16* dp = dst + r * din + half * hh;
  const float* gp = g + half * hh;
  const float* bp = bb + half * hh;
  for (int j = 0; j < hh; ++j) dp[j] = (_Float16)((p[j] - mu) * rs * gp[j] + bp[j]);
}

// ------------------------------------------------------------------
// SpMM diffusion: X[r] += v * emb[c]  (COO scatter)
// ------------------------------------------------------------------
__global__ void k_spmm(const int* __restrict__ rows, const int* __restrict__ cols,
                       const float* __restrict__ vals, const float* __restrict__ emb,
                       float* __restrict__ X, int nnz) {
  int i = blockIdx.x * 256 + threadIdx.x;
  if (i >= nnz) return;
  int r = rows[i], c = cols[i];
  float v = vals[i];
  const float4* e = (const float4*)(emb + (size_t)c * 64);
  float* xp = X + (size_t)r * 64;
#pragma unroll
  for (int j = 0; j < 16; ++j) {
    float4 t = e[j];
    atomicAdd(xp + j * 4 + 0, v * t.x);
    atomicAdd(xp + j * 4 + 1, v * t.y);
    atomicAdd(xp + j * 4 + 2, v * t.z);
    atomicAdd(xp + j * 4 + 3, v * t.w);
  }
}
__global__ void k_scatter_rows(const float* __restrict__ Y, const int* __restrict__ src,
                               const int* __restrict__ dst, float* __restrict__ S, int m) {
  int i = blockIdx.x * 256 + threadIdx.x;
  if (i >= m) return;
  const float4* yp = (const float4*)(Y + (size_t)src[i] * 64);
  float* d = S + (size_t)dst[i] * 64;
#pragma unroll
  for (int j = 0; j < 16; ++j) {
    float4 t = yp[j];
    atomicAdd(d + j * 4 + 0, t.x);
    atomicAdd(d + j * 4 + 1, t.y);
    atomicAdd(d + j * 4 + 2, t.z);
    atomicAdd(d + j * 4 + 3, t.w);
  }
}
__global__ void k_count(const int* __restrict__ idx, float* __restrict__ cnt, int m) {
  int i = blockIdx.x * 256 + threadIdx.x;
  if (i < m) atomicAdd(cnt + idx[i], 1.f);
}
__global__ void k_div(float* __restrict__ S, const float* __restrict__ cnt, int total) {
  int i = blockIdx.x * 256 + threadIdx.x;
  if (i < total) S[i] /= fmaxf(cnt[i >> 6], 1.f);
}
__global__ void k_combine(const float* __restrict__ XvS, const float* __restrict__ XvC,
                          const float* __restrict__ X0, float* __restrict__ Z, int total) {
  int i = blockIdx.x * 256 + threadIdx.x;
  if (i < total) Z[i] = 0.5f * (XvS[i] / fmaxf(XvC[i >> 6], 1.f)) + 0.5f * X0[i];
}

// ------------------------------------------------------------------
// AllSet MLP (din=64): out = W2 @ LN1(relu(W1 @ LN0(x) + b1)) + b2
// 128 threads = 4 waves, 16 rows/wave, matmuls via v_wmma.
// Activation tile async-DMA'd global->LDS; weights cvt to f16 once.
// ------------------------------------------------------------------
__global__ void k_mlp64(const float* __restrict__ X, float* __restrict__ out, int R,
                        const float* ln0g, const float* ln0b, const float* w1, const float* b1,
                        const float* ln1g, const float* ln1b, const float* w2, const float* b2) {
  __shared__ _Float16 w1h[64 * 64];
  __shared__ _Float16 w2h[64 * 64];
  __shared__ float raw[4][16 * 64];
  __shared__ _Float16 xt[4][16 * 64];
  __shared__ float ht[4][16 * 64];
  __shared__ _Float16 h2[4][16 * 64];
  int t = threadIdx.x, w = t >> 5, lane = t & 31;
  int rowbase = (blockIdx.x * 4 + w) * 16;
  bool act = rowbase < R;
  int rb = act ? rowbase : 0;
  // async DMA: 16 contiguous rows (4KB) straight into LDS
  async_stage_f32(X + (size_t)rb * 64, &raw[w][0], 16 * 64, lane, 32);
  // meanwhile convert weights to f16 in LDS via VGPR path
  for (int i = t; i < 64 * 64; i += 128) { w1h[i] = (_Float16)w1[i]; w2h[i] = (_Float16)w2[i]; }
  wait_async0();
  __syncthreads();
  ln_tile(raw[w], xt[w], 64, ln0g, ln0b, lane);
  __syncthreads();
  v8f acc[4];
  zero4(acc);
  gemm16(xt[w], w1h, 64, 0, lane, acc);
  {
    int col = lane & 15, rh = (lane & 16) >> 1;
#pragma unroll
    for (int nt = 0; nt < 4; ++nt) {
      float bv = b1[nt * 16 + col];
#pragma unroll
      for (int e = 0; e < 8; ++e) ht[w][(rh + e) * 64 + nt * 16 + col] = fmaxf(acc[nt][e] + bv, 0.f);
    }
  }
  __syncthreads();
  ln_tile(ht[w], h2[w], 64, ln1g, ln1b, lane);
  __syncthreads();
  v8f a2[4];
  zero4(a2);
  gemm16(h2[w], w2h, 64, 0, lane, a2);
  if (act) {
    int col = lane & 15, rh = (lane & 16) >> 1;
#pragma unroll
    for (int nt = 0; nt < 4; ++nt) {
      float bv = b2[nt * 16 + col];
#pragma unroll
      for (int e = 0; e < 8; ++e)
        out[(size_t)(rowbase + rh + e) * 64 + nt * 16 + col] = a2[nt][e] + bv;
    }
  }
}

// ------------------------------------------------------------------
// Fused m2: per incidence m: mlp2(concat(X[vertex[m]], Xe[edges[m]]))
// gather done with per-lane async-to-LDS DMA (dominant traffic of the
// whole net: 400k x 512B); scatter-accumulated into XvS[vertex[m]]
// ------------------------------------------------------------------
__global__ void k_m2(const float* __restrict__ Xd, const float* __restrict__ Xe,
                     const int* __restrict__ vertex, const int* __restrict__ edges,
                     float* __restrict__ XvS,
                     const float* ln0g, const float* ln0b, const float* w1, const float* b1,
                     const float* ln1g, const float* ln1b, const float* w2, const float* b2) {
  __shared__ _Float16 w1h[64 * 128];
  __shared__ _Float16 w2h[64 * 64];
  __shared__ float raw[4][16 * 128];
  __shared__ _Float16 xt[4][16 * 128];
  __shared__ float ht[4][16 * 64];
  __shared__ _Float16 h2[4][16 * 64];
  int t = threadIdx.x, w = t >> 5, lane = t & 31;
  int rowbase = (blockIdx.x * 4 + w) * 16;
  // gathered async DMA: tile = 16 rows x 128 floats = 512 chunks of 16B
  unsigned int ldsbase = (unsigned int)(size_t)&raw[w][0];
  for (int c = lane; c < 512; c += 32) {
    int r = c >> 5;           // row in tile
    int o = (c & 31) * 4;     // float offset within 128-wide row
    int m = rowbase + r;
    const float* src = (o < 64) ? (Xd + (size_t)vertex[m] * 64 + o)
                                : (Xe + (size_t)edges[m] * 64 + (o - 64));
    async_chunk16(ldsbase + (unsigned int)(r * 128 + o) * 4u, src);
  }
  // meanwhile convert weights to f16 in LDS via VGPR path
  for (int i = t; i < 64 * 128; i += 128) w1h[i] = (_Float16)w1[i];
  for (int i = t; i < 64 * 64; i += 128) w2h[i] = (_Float16)w2[i];
  wait_async0();
  __syncthreads();
  ln_tile(raw[w], xt[w], 128, ln0g, ln0b, lane);
  __syncthreads();
  v8f acc[4];
  zero4(acc);
  gemm16(xt[w], w1h, 128, 0, lane, acc);
  {
    int col = lane & 15, rh = (lane & 16) >> 1;
#pragma unroll
    for (int nt = 0; nt < 4; ++nt) {
      float bv = b1[nt * 16 + col];
#pragma unroll
      for (int e = 0; e < 8; ++e) ht[w][(rh + e) * 64 + nt * 16 + col] = fmaxf(acc[nt][e] + bv, 0.f);
    }
  }
  __syncthreads();
  ln_tile(ht[w], h2[w], 64, ln1g, ln1b, lane);
  __syncthreads();
  v8f a2[4];
  zero4(a2);
  gemm16(h2[w], w2h, 64, 0, lane, a2);
  {
    int col = lane & 15, rh = (lane & 16) >> 1;
#pragma unroll
    for (int nt = 0; nt < 4; ++nt) {
      float bv = b2[nt * 16 + col];
#pragma unroll
      for (int e = 0; e < 8; ++e) {
        int m = rowbase + rh + e;
        atomicAdd(&XvS[(size_t)vertex[m] * 64 + nt * 16 + col], a2[nt][e] + bv);
      }
    }
  }
}

// ------------------------------------------------------------------
// Generic out = X[R,K] @ W[NW,K]^T + bias  (optional relu), WMMA tiles
// dynamic LDS: W (f16, converted once) + per-wave 16xK x-tile (f16)
// ------------------------------------------------------------------
__global__ void k_gemm(const float* __restrict__ X, const float* __restrict__ W,
                       const float* __restrict__ bias, float* __restrict__ out,
                       int K, int NW, int relu) {
  extern __shared__ _Float16 sm[];
  _Float16* Wh = sm;
  int t = threadIdx.x, w = t >> 5, lane = t & 31;
  _Float16* xt = sm + (size_t)NW * K + (size_t)w * 16 * K;
  for (int i = t; i < NW * K; i += 128) Wh[i] = (_Float16)W[i];
  int ksh = 31 - __clz(K);
  int rowbase = (blockIdx.x * 4 + w) * 16;
  for (int idx = lane; idx < 16 * K; idx += 32)
    xt[idx] = (_Float16)X[(size_t)(rowbase + (idx >> ksh)) * K + (idx & (K - 1))];
  __syncthreads();
  int col = lane & 15, rh = (lane & 16) >> 1;
  for (int n0 = 0; n0 < NW; n0 += 64) {
    v8f acc[4];
    zero4(acc);
    gemm16(xt, Wh, K, n0, lane, acc);
#pragma unroll
    for (int nt = 0; nt < 4; ++nt) {
      int n = n0 + nt * 16 + col;
      float bv = bias[n];
#pragma unroll
      for (int e = 0; e < 8; ++e) {
        float v = acc[nt][e] + bv;
        if (relu) v = fmaxf(v, 0.f);
        out[(size_t)(rowbase + rh + e) * NW + n] = v;
      }
    }
  }
}

// ------------------------------------------------------------------
// Attention (dh=8, H=8, S=200). one wave per query row; VALU + shuffles
// mode 0: encoder pad (question==2); 1: causal + shifted-resp pad; 2: none
// ------------------------------------------------------------------
__global__ void k_attn(const float* __restrict__ Q, const float* __restrict__ Kb,
                       const float* __restrict__ Vb, float* __restrict__ O,
                       const int* __restrict__ question, const int* __restrict__ response,
                       int mode) {
  __shared__ float sc[4][200];
  int t = threadIdx.x, w = t >> 5, lane = t & 31;
  int b = blockIdx.x >> 3, h = blockIdx.x & 7;
  int q = blockIdx.y * 4 + w;
  const float* qp = Q + ((size_t)b * cS + q) * 64 + h * 8;
  float qv[8];
#pragma unroll
  for (int i = 0; i < 8; ++i) qv[i] = qp[i];
  const float scale = 0.35355339059327373f;  // 1/sqrt(8)
  float mx = -3.4e38f;
  for (int j = lane; j < cS; j += 32) {
    const float* kp = Kb + ((size_t)b * cS + j) * 64 + h * 8;
    float s = 0.f;
#pragma unroll
    for (int i = 0; i < 8; ++i) s += qv[i] * kp[i];
    s *= scale;
    if (mode == 0) {
      if (question[b * cS + j] == 2) s += NEGB;
    } else if (mode == 1) {
      if (j > q) s += NEGB;
      int rs = (j == 0) ? 0 : response[b * cS + j - 1];
      if (rs == 2) s += NEGB;
    }
    sc[w][j] = s;
    mx = fmaxf(mx, s);
  }
  mx = wred_max(mx);
  float sum = 0.f, acc[8];
#pragma unroll
  for (int i = 0; i < 8; ++i) acc[i] = 0.f;
  for (int j = lane; j < cS; j += 32) {
    float p = expf(sc[w][j] - mx);
    sum += p;
    const float* vp = Vb + ((size_t)b * cS + j) * 64 + h * 8;
#pragma unroll
    for (int i = 0; i < 8; ++i) acc[i] += p * vp[i];
  }
  sum = wred_sum(sum);
#pragma unroll
  for (int i = 0; i < 8; ++i) acc[i] = wred_sum(acc[i]);
  if (lane == 0) {
    float* op = O + ((size_t)b * cS + q) * 64 + h * 8;
#pragma unroll
    for (int i = 0; i < 8; ++i) op[i] = acc[i] / sum;
  }
}

// out = LN(x + y) rows of 64; y may be null. wave per row.
__global__ void k_addln(const float* __restrict__ x, const float* __restrict__ y,
                        const float* __restrict__ g, const float* __restrict__ bb,
                        float* __restrict__ out) {
  int t = threadIdx.x, w = t >> 5, lane = t & 31;
  size_t r = (size_t)blockIdx.x * 4 + w;
  float v0 = x[r * 64 + lane] + (y ? y[r * 64 + lane] : 0.f);
  float v1 = x[r * 64 + lane + 32] + (y ? y[r * 64 + lane + 32] : 0.f);
  float s = wred_sum(v0 + v1);
  float q = wred_sum(v0 * v0 + v1 * v1);
  float mu = s / 64.f;
  float rs = rsqrtf(q / 64.f - mu * mu + EPSB);
  out[r * 64 + lane] = (v0 - mu) * rs * g[lane] + bb[lane];
  out[r * 64 + lane + 32] = (v1 - mu) * rs * g[lane + 32] + bb[lane + 32];
}

static __device__ __forceinline__ float pe_val(int s, int d) {
  int i = d >> 1;
  float div = expf(-(float)(2 * i) * (9.210340371976184f / 64.0f));
  float ang = (float)s * div;
  return (d & 1) ? cosf(ang) : sinf(ang);
}
__global__ void k_build_eq(const float* __restrict__ conv, const int* __restrict__ question,
                           float* __restrict__ eq) {
  int i = blockIdx.x * 256 + threadIdx.x;  // B*S*64
  int d = i & 63, bs = i >> 6;
  eq[i] = conv[(size_t)question[bs] * 64 + d] + pe_val(bs % cS, d);
}
__global__ void k_build_er(const float* __restrict__ emb_resp, const int* __restrict__ response,
                           float* __restrict__ er) {
  int i = blockIdx.x * 256 + threadIdx.x;
  int d = i & 63, bs = i >> 6, s = bs % cS;
  int r = (s == 0) ? 0 : response[bs - 1];
  er[i] = emb_resp[r * 64 + d] + pe_val(s, d);
}
__global__ void k_pred(const float* __restrict__ Y, const float* __restrict__ pw,
                       const float* __restrict__ pb, float* __restrict__ out) {
  int t = threadIdx.x, w = t >> 5, lane = t & 31;
  size_t r = (size_t)blockIdx.x * 4 + w;
  float s = Y[r * 64 + lane] * pw[lane] + Y[r * 64 + lane + 32] * pw[lane + 32];
  s = wred_sum(s);
  if (lane == 0) out[r] = 1.f / (1.f + expf(-(s + pb[0])));
}

// ------------------------------------------------------------------
extern "C" void kernel_launch(void* const* d_in, const int* in_sizes, int n_in,
                              void* d_out, int out_size, void* d_ws, size_t ws_size,
                              hipStream_t stream) {
  (void)in_sizes; (void)n_in; (void)out_size; (void)ws_size;
  const int* question = (const int*)d_in[0];
  const int* response = (const int*)d_in[1];
  const int* edge_index = (const int*)d_in[3];
  const int* vertex = edge_index;
  const int* edges = edge_index + cM;
  const int* dif_rows = (const int*)d_in[4];
  const int* dif_cols = (const int*)d_in[5];
  const float* dif_vals = (const float*)d_in[6];
  const float* emb_node = (const float*)d_in[7];
  const float* emb_resp = (const float*)d_in[8];
  const float* m1p[8], *m2p[8], *m3p[8];
  for (int i = 0; i < 8; ++i) { m1p[i] = (const float*)d_in[9 + i]; m2p[i] = (const float*)d_in[17 + i]; m3p[i] = (const float*)d_in[25 + i]; }
  const float* enc_qkv_w = (const float*)d_in[33]; const float* enc_qkv_b = (const float*)d_in[34];
  const float* enc_out_w = (const float*)d_in[35]; const float* enc_out_b = (const float*)d_in[36];
  const float* enc_ff1_w = (const float*)d_in[37]; const float* enc_ff1_b = (const float*)d_in[38];
  const float* enc_ff2_w = (const float*)d_in[39]; const float* enc_ff2_b = (const float*)d_in[40];
  const float* enc_ln1_g = (const float*)d_in[41]; const float* enc_ln1_b = (const float*)d_in[42];
  const float* enc_ln2_g = (const float*)d_in[43]; const float* enc_ln2_b = (const float*)d_in[44];
  const float* enc_norm_g = (const float*)d_in[45]; const float* enc_norm_b = (const float*)d_in[46];
  const float* dec_sa_qkv_w = (const float*)d_in[47]; const float* dec_sa_qkv_b = (const float*)d_in[48];
  const float* dec_sa_out_w = (const float*)d_in[49]; const float* dec_sa_out_b = (const float*)d_in[50];
  const float* dec_ca_qkv_w = (const float*)d_in[51]; const float* dec_ca_qkv_b = (const float*)d_in[52];
  const float* dec_ca_out_w = (const float*)d_in[53]; const float* dec_ca_out_b = (const float*)d_in[54];
  const float* dec_ff1_w = (const float*)d_in[55]; const float* dec_ff1_b = (const float*)d_in[56];
  const float* dec_ff2_w = (const float*)d_in[57]; const float* dec_ff2_b = (const float*)d_in[58];
  const float* dec_ln1_g = (const float*)d_in[59]; const float* dec_ln1_b = (const float*)d_in[60];
  const float* dec_ln2_g = (const float*)d_in[61]; const float* dec_ln2_b = (const float*)d_in[62];
  const float* dec_ln3_g = (const float*)d_in[63]; const float* dec_ln3_b = (const float*)d_in[64];
  const float* dec_norm_g = (const float*)d_in[65]; const float* dec_norm_b = (const float*)d_in[66];
  const float* pred_w = (const float*)d_in[67]; const float* pred_b = (const float*)d_in[68];

  // workspace carve-up (~117 MB of f32)
  float* ws = (float*)d_ws;
  size_t o = 0;
  auto alloc = [&](size_t n) { float* p = ws + o; o += n; return p; };
  float* Xd  = alloc((size_t)cN * 64);
  float* Y1  = alloc((size_t)cN * 64);
  float* XeS = alloc((size_t)cNUME * 64);
  float* XeC = alloc(cNUME);
  float* XvS = alloc((size_t)cN * 64);
  float* XvC = alloc(cN);
  float* Zc  = alloc((size_t)cN * 64);
  float* cnv = alloc((size_t)cN * 64);
  float* X1  = alloc((size_t)cBS * 64);
  float* Yd  = alloc((size_t)cBS * 64);
  float* bQ  = alloc((size_t)cBS * 64);
  float* bK  = alloc((size_t)cBS * 64);
  float* bV  = alloc((size_t)cBS * 64);
  float* bA  = alloc((size_t)cBS * 64);
  float* bO  = alloc((size_t)cBS * 64);
  float* bM  = alloc((size_t)cBS * 64);
  float* bF  = alloc((size_t)cBS * 512);

  hipMemsetAsync(Xd, 0, (size_t)cN * 64 * 4, stream);
  hipMemsetAsync(XeS, 0, (size_t)cNUME * 64 * 4, stream);
  hipMemsetAsync(XeC, 0, (size_t)cNUME * 4, stream);
  hipMemsetAsync(XvS, 0, (size_t)cN * 64 * 4, stream);
  hipMemsetAsync(XvC, 0, (size_t)cN * 4, stream);

  // ---- hypergraph conv ----
  k_spmm<<<(cNNZ + 255) / 256, 256, 0, stream>>>(dif_rows, dif_cols, dif_vals, emb_node, Xd, cNNZ);
  int mlpBlocks = ((cN / 16) + 3) / 4;
  k_mlp64<<<mlpBlocks, 128, 0, stream>>>(Xd, Y1, cN, m1p[0], m1p[1], m1p[2], m1p[3], m1p[4], m1p[5], m1p[6], m1p[7]);
  k_scatter_rows<<<(cM + 255) / 256, 256, 0, stream>>>(Y1, vertex, edges, XeS, cM);
  k_count<<<(cM + 255) / 256, 256, 0, stream>>>(edges, XeC, cM);
  k_div<<<(cNUME * 64 + 255) / 256, 256, 0, stream>>>(XeS, XeC, cNUME * 64);  // XeS -> Xe
  k_m2<<<cM / 64, 128, 0, stream>>>(Xd, XeS, vertex, edges, XvS,
                                    m2p[0], m2p[1], m2p[2], m2p[3], m2p[4], m2p[5], m2p[6], m2p[7]);
  k_count<<<(cM + 255) / 256, 256, 0, stream>>>(vertex, XvC, cM);
  k_combine<<<(cN * 64 + 255) / 256, 256, 0, stream>>>(XvS, XvC, Xd, Zc, cN * 64);
  k_mlp64<<<mlpBlocks, 128, 0, stream>>>(Zc, cnv, cN, m3p[0], m3p[1], m3p[2], m3p[3], m3p[4], m3p[5], m3p[6], m3p[7]);

  // ---- sequence embeddings ----
  k_build_eq<<<(cBS * 64) / 256, 256, 0, stream>>>(cnv, question, X1);
  k_build_er<<<(cBS * 64) / 256, 256, 0, stream>>>(emb_resp, response, Yd);

  auto gemm = [&](const float* X, const float* W, const float* bias, float* out, int K, int NW, int relu) {
    size_t sh = 2 * ((size_t)NW * K + (size_t)64 * K);
    k_gemm<<<cBS / 64, 128, sh, stream>>>(X, W, bias, out, K, NW, relu);
  };
  dim3 agrid(cB * 8, cS / 4);

  // ---- encoder ----
  for (int i = 0; i < 2; ++i) {
    const float* qw = enc_qkv_w + (size_t)i * 192 * 64;
    const float* qb = enc_qkv_b + (size_t)i * 192;
    gemm(X1, qw, qb, bQ, 64, 64, 0);
    gemm(X1, qw + 64 * 64, qb + 64, bK, 64, 64, 0);
    gemm(X1, qw + 128 * 64, qb + 128, bV, 64, 64, 0);
    k_attn<<<agrid, 128, 0, stream>>>(bQ, bK, bV, bA, question, response, 0);
    gemm(bA, enc_out_w + (size_t)i * 64 * 64, enc_out_b + i * 64, bO, 64, 64, 0);
    k_addln<<<cBS / 4, 128, 0, stream>>>(X1, bO, enc_ln1_g + i * 64, enc_ln1_b + i * 64, X1);
    gemm(X1, enc_ff1_w + (size_t)i * 512 * 64, enc_ff1_b + i * 512, bF, 64, 512, 1);
    gemm(bF, enc_ff2_w + (size_t)i * 64 * 512, enc_ff2_b + i * 64, bO, 512, 64, 0);
    k_addln<<<cBS / 4, 128, 0, stream>>>(X1, bO, enc_ln2_g + i * 64, enc_ln2_b + i * 64, X1);
  }
  k_addln<<<cBS / 4, 128, 0, stream>>>(X1, nullptr, enc_norm_g, enc_norm_b, bM);

  // ---- decoder ----
  for (int i = 0; i < 2; ++i) {
    const float* sw = dec_sa_qkv_w + (size_t)i * 192 * 64;
    const float* sb = dec_sa_qkv_b + (size_t)i * 192;
    gemm(Yd, sw, sb, bQ, 64, 64, 0);
    gemm(Yd, sw + 64 * 64, sb + 64, bK, 64, 64, 0);
    gemm(Yd, sw + 128 * 64, sb + 128, bV, 64, 64, 0);
    k_attn<<<agrid, 128, 0, stream>>>(bQ, bK, bV, bA, question, response, 1);
    gemm(bA, dec_sa_out_w + (size_t)i * 64 * 64, dec_sa_out_b + i * 64, bO, 64, 64, 0);
    k_addln<<<cBS / 4, 128, 0, stream>>>(Yd, bO, dec_ln1_g + i * 64, dec_ln1_b + i * 64, Yd);
    const float* cw = dec_ca_qkv_w + (size_t)i * 192 * 64;
    const float* cb = dec_ca_qkv_b + (size_t)i * 192;
    gemm(Yd, cw, cb, bQ, 64, 64, 0);
    gemm(bM, cw + 64 * 64, cb + 64, bK, 64, 64, 0);
    gemm(bM, cw + 128 * 64, cb + 128, bV, 64, 64, 0);
    k_attn<<<agrid, 128, 0, stream>>>(bQ, bK, bV, bA, question, response, 2);
    gemm(bA, dec_ca_out_w + (size_t)i * 64 * 64, dec_ca_out_b + i * 64, bO, 64, 64, 0);
    k_addln<<<cBS / 4, 128, 0, stream>>>(Yd, bO, dec_ln2_g + i * 64, dec_ln2_b + i * 64, Yd);
    gemm(Yd, dec_ff1_w + (size_t)i * 512 * 64, dec_ff1_b + i * 512, bF, 64, 512, 1);
    gemm(bF, dec_ff2_w + (size_t)i * 64 * 512, dec_ff2_b + i * 64, bO, 512, 64, 0);
    k_addln<<<cBS / 4, 128, 0, stream>>>(Yd, bO, dec_ln3_g + i * 64, dec_ln3_b + i * 64, Yd);
  }
  k_addln<<<cBS / 4, 128, 0, stream>>>(Yd, nullptr, dec_norm_g, dec_norm_b, Yd);
  k_pred<<<cBS / 4, 128, 0, stream>>>(Yd, pred_w, pred_b, (float*)d_out);
}
